// TimeAttention_74792560492702
// MI455X (gfx1250) — matrix operand order repaired and checked by reading, not compiled, not run on hardware
//
#include <hip/hip_runtime.h>

// TimeAttention for gfx1250 (MI455X): flash-attention on v_wmma_f32_16x16x32_f16,
// computed transposed (S^T = K.Q^T, out^T = V^T.P^T) so softmax statistics are
// per-lane scalars. Causal masking touches only the per-var-block diagonal tile.
// Shapes fixed by the reference: B=8, L=1024, H=8, E=64, n_vars=8, n_tokens=128.

constexpr int B_  = 8;
constexpr int L_  = 1024;
constexpr int H_  = 8;
constexpr int E_  = 64;
constexpr int NT_ = 128;               // n_tokens
constexpr int NV_ = 8;                 // n_vars
constexpr float SCALE   = 0.125f;      // 1/sqrt(64)
constexpr float LOG2E   = 1.44269504088896341f;
constexpr float NEG_BIG = -1.0e30f;
// log2(10000)/16 : theta_j = 10000^(-j/16) = 2^(-j * 0.83048...)
constexpr float ROPE_L2 = 0.83048202372184057f;

constexpr size_t KH_ELEMS = (size_t)B_ * H_ * L_ * E_;   // 4M halves = 8MB
constexpr size_t WS_NEED  = 2 * KH_ELEMS * sizeof(_Float16);

typedef _Float16 half16 __attribute__((ext_vector_type(16)));
typedef _Float16 half8v __attribute__((ext_vector_type(8)));
typedef float    float8 __attribute__((ext_vector_type(8)));

static __device__ __forceinline__ float8 wmma_f16(half16 a, half16 b, float8 c) {
  // D = A(16x32 f16) * B(32x16 f16) + C(16x16 f32)
  return __builtin_amdgcn_wmma_f32_16x16x32_f16(false, a, false, b, (short)0, c,
                                                false, false);
}

static __device__ __forceinline__ void load8(float* t, const float* __restrict__ p) {
  const float4* p4 = (const float4*)p;
#pragma unroll
  for (int i = 0; i < 2; ++i) {
    float4 v = p4[i];
    t[4 * i + 0] = v.x; t[4 * i + 1] = v.y; t[4 * i + 2] = v.z; t[4 * i + 3] = v.w;
  }
}
static __device__ __forceinline__ void load16(float* t, const float* __restrict__ p) {
  const float4* p4 = (const float4*)p;
#pragma unroll
  for (int i = 0; i < 4; ++i) {
    float4 v = p4[i];
    t[4 * i + 0] = v.x; t[4 * i + 1] = v.y; t[4 * i + 2] = v.z; t[4 * i + 3] = v.w;
  }
}

// Rotate 8 consecutive head-dims starting at even index e0 (RoPE region e<32).
static __device__ __forceinline__ void rope8(float* x, int e0, float pos) {
#pragma unroll
  for (int i = 0; i < 4; ++i) {
    float j = (float)((e0 >> 1) + i);                       // pair index = e/2
    float theta = __builtin_amdgcn_exp2f(-j * ROPE_L2);     // 10000^(-j/16)
    float ang = pos * theta;
    float sn, cs;
    __sincosf(ang, &sn, &cs);
    float a = x[2 * i], b = x[2 * i + 1];
    x[2 * i]     = cs * a - sn * b;
    x[2 * i + 1] = cs * b + sn * a;
  }
}

// ---------------- Phase 1: K RoPE->f16, V transpose->f16 -------------------
__global__ __launch_bounds__(32)
void preproc_kernel(const float* __restrict__ K, const float* __restrict__ V,
                    _Float16* __restrict__ Kh, _Float16* __restrict__ Vt) {
  __shared__ alignas(16) _Float16 vt[32][E_];   // 4KB V tile for transpose

  const int lane = threadIdx.x;
  const int tile = blockIdx.x % (L_ / 32);
  const int bh   = blockIdx.x / (L_ / 32);
  const int b    = bh / H_;
  const int h    = bh % H_;
  const int s    = tile * 32 + lane;            // one l-row per lane
  const float pos = (float)(s & (NT_ - 1));

  const size_t rs = (size_t)H_ * E_;

  // ---- K row: load f32, RoPE e<32, store 64 contiguous halves
  {
    const float* kp = K + ((size_t)b * L_ + s) * rs + (size_t)h * E_;
    float t[64];
    load16(t, kp); load16(t + 16, kp + 16); load16(t + 32, kp + 32); load16(t + 48, kp + 48);
    rope8(t, 0, pos); rope8(t + 8, 8, pos); rope8(t + 16, 16, pos); rope8(t + 24, 24, pos);
    _Float16* kr = Kh + ((size_t)bh * L_ + s) * E_;
#pragma unroll
    for (int c = 0; c < 4; ++c) {
      half16 hv;
#pragma unroll
      for (int i = 0; i < 16; ++i) hv[i] = (_Float16)t[16 * c + i];
      *(half16*)(kr + 16 * c) = hv;
    }
  }

  // ---- V row -> LDS (f16), then column-wise write-out: Vt[bh][e][s]
  {
    const float* vp = V + ((size_t)b * L_ + s) * rs + (size_t)h * E_;
    float t[64];
    load16(t, vp); load16(t + 16, vp + 16); load16(t + 32, vp + 32); load16(t + 48, vp + 48);
#pragma unroll
    for (int c = 0; c < 4; ++c) {
      half16 hv;
#pragma unroll
      for (int i = 0; i < 16; ++i) hv[i] = (_Float16)t[16 * c + i];
      *(half16*)&vt[lane][16 * c] = hv;
    }
  }
  asm volatile("s_wait_dscnt 0" ::: "memory");  // cross-lane LDS visibility

  // lane owns e-pair (2*lane, 2*lane+1); u32 column reads are bank-conflict-free
  {
    const int e0 = 2 * lane;
    _Float16 ra[32], rb[32];
#pragma unroll
    for (int si = 0; si < 32; ++si) {
      union { unsigned u; _Float16 hh[2]; } cv;
      cv.u = *(const unsigned*)&vt[si][e0];
      ra[si] = cv.hh[0];
      rb[si] = cv.hh[1];
    }
    _Float16* oa = Vt + ((size_t)bh * E_ + e0) * L_ + tile * 32;
    _Float16* ob = oa + L_;
#pragma unroll
    for (int c = 0; c < 2; ++c) {
      half16 ha, hb;
#pragma unroll
      for (int i = 0; i < 16; ++i) { ha[i] = ra[16 * c + i]; hb[i] = rb[16 * c + i]; }
      *(half16*)(oa + 16 * c) = ha;
      *(half16*)(ob + 16 * c) = hb;
    }
  }
}

// ---------------- Phase 2: flash attention (transposed tiles) --------------
template <bool PRE>
__global__ __launch_bounds__(32)
void time_attn_wmma_kernel(const float* __restrict__ Q, const float* __restrict__ K,
                           const float* __restrict__ V, const float* __restrict__ EW,
                           const _Float16* __restrict__ Kh,
                           const _Float16* __restrict__ Vt, float* __restrict__ O) {
  __shared__ alignas(16) _Float16 pbuf[16 * 32];   // P^T staging [q][s], 1KB
  __shared__ alignas(16) float    obuf[16][68];    // out^T un-transpose, padded

  const int lane = threadIdx.x;           // 0..31 (wave32)
  const int g    = lane >> 4;             // half-wave group
  const int ln   = lane & 15;
  const int g8   = g * 8;

  const int qtile = blockIdx.x & (L_ / 16 - 1);
  const int bh    = blockIdx.x / (L_ / 16);
  const int b     = bh / H_;
  const int h     = bh % H_;
  const int q0    = qtile * 16;
  const int pq0   = q0 & (NT_ - 1);
  const int varq  = q0 / NT_;
  const int pqln  = pq0 + ln;             // this lane's query token position

  const float w0s = EW[h] * SCALE;        // emb_weight[0][h] * scale
  const float w1s = EW[H_ + h] * SCALE;   // emb_weight[1][h] * scale

  const size_t rs = (size_t)H_ * E_;
  const float* qbase = Q + (size_t)b * L_ * rs + (size_t)h * E_;
  const float* kbase = K + (size_t)b * L_ * rs + (size_t)h * E_;
  const float* vbase = V + (size_t)b * L_ * rs + (size_t)h * E_;

  // ---- Q^T B-fragments: lane = query column q0+ln, halves = two 8-wide e-runs
  half16 bQ0, bQ1;
  {
    const float* qp = qbase + (size_t)(q0 + ln) * rs;
    const float pql = (float)pqln;
    float t[16];
    load8(t, qp + g8);                      // e = g8.., 16+g8.. (RoPE region)
    load8(t + 8, qp + 16 + g8);
    rope8(t, g8, pql);
    rope8(t + 8, 16 + g8, pql);
#pragma unroll
    for (int i = 0; i < 16; ++i) bQ0[i] = (_Float16)(t[i] * SCALE);
    load8(t, qp + 32 + g8);                 // e = 32+g8.., 48+g8.. (no RoPE)
    load8(t + 8, qp + 48 + g8);
#pragma unroll
    for (int i = 0; i < 16; ++i) bQ1[i] = (_Float16)(t[i] * SCALE);
  }

  // ---- loop-invariant diagonal-tile keep masks (hoisted by the compiler)
  const int diag = pq0 & ~31;             // in-block offset of the partial tile
  const int thr0 = (pq0 & 31) + ln - g8;  // keep iff subtile_row <= thr0 - 16j
  bool keep0[8], keep1[8];
#pragma unroll
  for (int r = 0; r < 8; ++r) {
    keep0[r] = (r <= thr0);
    keep1[r] = (r <= thr0 - 16);
  }

  float mcur = NEG_BIG, lcur = 0.0f;      // per-lane (per-query) statistics
  float8 accT[4];
#pragma unroll
  for (int j = 0; j < 4; ++j)
#pragma unroll
    for (int r = 0; r < 8; ++r) accT[j][r] = 0.0f;

  const float8 zero8 = {0.f, 0.f, 0.f, 0.f, 0.f, 0.f, 0.f, 0.f};

  for (int vb = 0; vb < NV_; ++vb) {
    const float biasv = (vb == varq) ? w1s : w0s;
    for (int t = 0; t <= diag; t += 32) {       // tiles past diag are fully masked
      const int s0 = vb * NT_ + t;

      // ---- K A-fragments (lane = s-row, halves = two 8-wide e-runs)
      half16 ak[2][2];
#pragma unroll
      for (int j = 0; j < 2; ++j) {
        const int s = s0 + 16 * j + ln;
        if constexpr (PRE) {
          const _Float16* kr = Kh + ((size_t)bh * L_ + s) * E_;
          half8v lo0 = *(const half8v*)(kr + g8);
          half8v hi0 = *(const half8v*)(kr + 16 + g8);
          half8v lo1 = *(const half8v*)(kr + 32 + g8);
          half8v hi1 = *(const half8v*)(kr + 48 + g8);
#pragma unroll
          for (int i = 0; i < 8; ++i) {
            ak[j][0][i] = lo0[i]; ak[j][0][8 + i] = hi0[i];
            ak[j][1][i] = lo1[i]; ak[j][1][8 + i] = hi1[i];
          }
        } else {
          const float ps = (float)(t + 16 * j + ln);
          const float* kp = kbase + (size_t)s * rs;
          float u[8];
          load8(u, kp + g8);           rope8(u, g8, ps);
#pragma unroll
          for (int i = 0; i < 8; ++i) ak[j][0][i] = (_Float16)u[i];
          load8(u, kp + 16 + g8);      rope8(u, 16 + g8, ps);
#pragma unroll
          for (int i = 0; i < 8; ++i) ak[j][0][8 + i] = (_Float16)u[i];
          load8(u, kp + 32 + g8);
#pragma unroll
          for (int i = 0; i < 8; ++i) ak[j][1][i] = (_Float16)u[i];
          load8(u, kp + 48 + g8);
#pragma unroll
          for (int i = 0; i < 8; ++i) ak[j][1][8 + i] = (_Float16)u[i];
        }
      }

      // ---- scores S^T: D(s_local, q) per subtile; K-dim chained over e
      float8 c0 = wmma_f16(ak[0][0], bQ0, zero8);
      c0        = wmma_f16(ak[0][1], bQ1, c0);
      float8 c1 = wmma_f16(ak[1][0], bQ0, zero8);
      c1        = wmma_f16(ak[1][1], bQ1, c1);

      // ---- raw (unbiased) scores; mask only on the diagonal tile (uniform br)
      float cm[16];
      if (t == diag) {
#pragma unroll
        for (int r = 0; r < 8; ++r) {
          cm[r]     = keep0[r] ? c0[r] : NEG_BIG;
          cm[8 + r] = keep1[r] ? c1[r] : NEG_BIG;
        }
      } else {
#pragma unroll
        for (int r = 0; r < 8; ++r) { cm[r] = c0[r]; cm[8 + r] = c1[r]; }
      }

      // ---- per-lane online softmax; bias folded into the exp shift
      float tmax = cm[0];
#pragma unroll
      for (int i = 1; i < 16; ++i) tmax = fmaxf(tmax, cm[i]);
      tmax = fmaxf(tmax, __shfl_xor(tmax, 16, 32));
      const float mnew = fmaxf(mcur, tmax + biasv);
      const float al = __builtin_amdgcn_exp2f((mcur - mnew) * LOG2E);
      const float mshift = (mnew - biasv) * LOG2E;   // p = exp2(c*log2e - mshift)
      float psum = 0.0f;
#pragma unroll
      for (int i = 0; i < 16; ++i) {
        const float p = __builtin_amdgcn_exp2f(fmaf(cm[i], LOG2E, -mshift));
        cm[i] = p;
        psum += p;
      }
      psum += __shfl_xor(psum, 16, 32);
      lcur = lcur * al + psum;
      mcur = mnew;
#pragma unroll
      for (int j = 0; j < 4; ++j)
#pragma unroll
        for (int r = 0; r < 8; ++r) accT[j][r] *= al;

      // ---- stage P^T in LDS as [q][s], read back as two 8-wide s-runs B-fragment
#pragma unroll
      for (int r = 0; r < 8; ++r) {
        pbuf[ln * 32 + r + g8]      = (_Float16)cm[r];
        pbuf[ln * 32 + 16 + r + g8] = (_Float16)cm[8 + r];
      }
      asm volatile("s_wait_dscnt 0" ::: "memory");  // cross-lane store->load fence
      half16 bP;
      {
        const half8v plo = *(const half8v*)&pbuf[ln * 32 + g8];
        const half8v phi = *(const half8v*)&pbuf[ln * 32 + 16 + g8];
#pragma unroll
        for (int i = 0; i < 8; ++i) { bP[i] = plo[i]; bP[8 + i] = phi[i]; }
      }

      // ---- out^T += V^T . P^T  (A = V^T e-subtile, two 8-wide s-runs per lane)
#pragma unroll
      for (int je = 0; je < 4; ++je) {
        half16 av;
        if constexpr (PRE) {
          const _Float16* vr = Vt + ((size_t)bh * E_ + 16 * je + ln) * L_ + s0;
          half8v lo = *(const half8v*)(vr + g8);
          half8v hi = *(const half8v*)(vr + 16 + g8);
#pragma unroll
          for (int i = 0; i < 8; ++i) { av[i] = lo[i]; av[8 + i] = hi[i]; }
        } else {
          const float* vp = vbase + (size_t)s0 * rs + 16 * je + ln;
#pragma unroll
          for (int i = 0; i < 8; ++i) {
            av[i]     = (_Float16)vp[(size_t)(g8 + i) * rs];
            av[8 + i] = (_Float16)vp[(size_t)(16 + g8 + i) * rs];
          }
        }
        accT[je] = wmma_f16(av, bP, accT[je]);
      }
    }
  }

  // ---- epilogue: normalize, un-transpose through padded LDS, coalesced store
  const float oinv = 1.0f / lcur;
#pragma unroll
  for (int je = 0; je < 4; ++je)
#pragma unroll
    for (int r = 0; r < 8; ++r)
      obuf[ln][16 * je + r + g8] = accT[je][r] * oinv;   // [q][e]
  asm volatile("s_wait_dscnt 0" ::: "memory");
#pragma unroll
  for (int i = 0; i < 8; ++i) {
    const int row = 2 * i + g;                            // query row in tile
    float4 vv = *(const float4*)&obuf[row][4 * ln];
    *(float4*)(O + ((size_t)b * L_ + q0 + row) * rs + (size_t)h * E_ + 4 * ln) = vv;
  }
}

extern "C" void kernel_launch(void* const* d_in, const int* in_sizes, int n_in,
                              void* d_out, int out_size, void* d_ws, size_t ws_size,
                              hipStream_t stream) {
  (void)in_sizes; (void)n_in; (void)out_size;
  const float* q  = (const float*)d_in[0];
  const float* k  = (const float*)d_in[1];
  const float* v  = (const float*)d_in[2];
  const float* ew = (const float*)d_in[3];
  float* out = (float*)d_out;

  dim3 block(32);
  dim3 grid_attn(B_ * H_ * (L_ / 16));            // 4096 single-wave workgroups

  if (ws_size >= WS_NEED && d_ws != nullptr) {
    _Float16* Kh = (_Float16*)d_ws;
    _Float16* Vt = Kh + KH_ELEMS;
    dim3 grid_pre(B_ * H_ * (L_ / 32));            // 2048 blocks
    preproc_kernel<<<grid_pre, block, 0, stream>>>(k, v, Kh, Vt);
    time_attn_wmma_kernel<true><<<grid_attn, block, 0, stream>>>(q, k, v, ew, Kh, Vt, out);
  } else {
    time_attn_wmma_kernel<false><<<grid_attn, block, 0, stream>>>(q, k, v, ew,
                                                                  nullptr, nullptr, out);
  }
}
